// BDH_GPU_57578331570195
// MI455X (gfx1250) — compile-verified, matrix operand order checked by reading, not run on hardware
//
#include <hip/hip_runtime.h>

// ---------------- types ----------------
typedef __bf16 bf16;
typedef __attribute__((ext_vector_type(16))) __bf16 v16bf;
typedef __attribute__((ext_vector_type(8)))  __bf16 v8bf;
typedef __attribute__((ext_vector_type(8)))  float  v8f;

// ---------------- problem constants ----------------
static constexpr int T_SEQ  = 1024;
static constexpr int D_DIM  = 256;
static constexpr int H_HEADS= 4;
static constexpr int NH     = 8192;     // N / H
static constexpr int N_DIM  = 32768;
static constexpr int V_DIM  = 256;
static constexpr int L_LAYERS = 6;
static constexpr float TWO_PI = 6.283185307179586f;
static constexpr float INV_2PI = 0.15915494309189535f;
static constexpr int LDS_STRIDE = 40;   // 32 K-elems + 8 pad (80B rows, 16B aligned)

// ---------------- WMMA helpers ----------------
__device__ __forceinline__ v8f wmma_bf16(v16bf a, v16bf b, v8f c) {
  return __builtin_amdgcn_wmma_f32_16x16x32_bf16(false, a, false, b, (short)0, c, false, false);
}

// Fragment load (16 rows x 32 K of bf16 from row-major memory, ld in elements).
// Lane r=lane&15 holds row row0+r; hi=lane>>4 selects K-halves {0..7,16..23} /
// {8..15,24..31}: two contiguous 16-byte loads per lane. B fragments use the
// same loader on pre-transposed operands.
__device__ __forceinline__ v16bf load_frag(const bf16* __restrict__ base, long ld,
                                           int row0, int k0, int lane) {
  const int r  = lane & 15;
  const int hi = lane >> 4;
  const bf16* p = base + (long)(row0 + r) * ld + (k0 + 8 * hi);
  v8bf lo = *(const v8bf*)(p);
  v8bf hh = *(const v8bf*)(p + 16);
  return __builtin_shufflevector(lo, hh, 0, 1, 2, 3, 4, 5, 6, 7,
                                         8, 9, 10, 11, 12, 13, 14, 15);
}

// Same fragment layout out of an LDS panel (64 rows x 32 K, stride LDS_STRIDE).
__device__ __forceinline__ v16bf lds_frag(const bf16* pan, int row0, int lane) {
  const int r  = lane & 15;
  const int hi = lane >> 4;
  const bf16* p = pan + (row0 + r) * LDS_STRIDE + 8 * hi;
  v8bf lo = *(const v8bf*)(p);
  v8bf hh = *(const v8bf*)(p + 16);
  return __builtin_shufflevector(lo, hh, 0, 1, 2, 3, 4, 5, 6, 7,
                                         8, 9, 10, 11, 12, 13, 14, 15);
}

__device__ __forceinline__ float wave_sum(float v) {
#pragma unroll
  for (int m = 16; m >= 1; m >>= 1) v += __shfl_xor(v, m, 32);
  return v;
}

// ---------------- setup kernels ----------------
// convert f32 -> bf16 with transpose: src (batch, rows, cols) -> dst (batch, cols, rows)
__global__ __launch_bounds__(256) void k_cvt_T(const float* __restrict__ s,
                                               bf16* __restrict__ d,
                                               int rows, int cols, long per_batch) {
  const long i = (long)blockIdx.x * 256 + threadIdx.x;
  if (i >= per_batch) return;
  const long b = blockIdx.y;
  const long r = i / cols;
  const long c = i - r * cols;
  d[b * per_batch + c * rows + r] = (bf16)s[b * per_batch + i];
}

// embedding gather + LayerNorm -> v (f32 master + bf16 + bf16-transposed)
__global__ __launch_bounds__(128) void k_embed(const int* __restrict__ idx,
                                               const float* __restrict__ wte,
                                               float* __restrict__ vf,
                                               bf16* __restrict__ vb,
                                               bf16* __restrict__ vbT) {
  const int lane = threadIdx.x & 31;
  const int t = blockIdx.x * 4 + (threadIdx.x >> 5);
  const int tok = idx[t];
  const float* p = wte + (long)tok * D_DIM;
  float v[8]; float s = 0.f;
#pragma unroll
  for (int i = 0; i < 8; ++i) { v[i] = p[i * 32 + lane]; s += v[i]; }
  const float mean = wave_sum(s) * (1.0f / D_DIM);
  float s2 = 0.f;
#pragma unroll
  for (int i = 0; i < 8; ++i) { float d = v[i] - mean; s2 += d * d; }
  const float rstd = rsqrtf(wave_sum(s2) * (1.0f / D_DIM) + 1e-5f);
#pragma unroll
  for (int i = 0; i < 8; ++i) {
    const int dcol = i * 32 + lane;
    float nv = (v[i] - mean) * rstd;
    vf[(long)t * D_DIM + dcol] = nv;
    vb[(long)t * D_DIM + dcol] = (bf16)nv;
    vbT[(long)dcol * T_SEQ + t] = (bf16)nv;
  }
}

// generic row LayerNorm (256 cols) f32 -> bf16
__global__ __launch_bounds__(128) void k_ln_rows(const float* __restrict__ src,
                                                 bf16* __restrict__ dst) {
  const int lane = threadIdx.x & 31;
  const long row = (long)blockIdx.x * 4 + (threadIdx.x >> 5);
  const float* p = src + row * D_DIM;
  float v[8]; float s = 0.f;
#pragma unroll
  for (int i = 0; i < 8; ++i) { v[i] = p[i * 32 + lane]; s += v[i]; }
  const float mean = wave_sum(s) * (1.0f / D_DIM);
  float s2 = 0.f;
#pragma unroll
  for (int i = 0; i < 8; ++i) { float d = v[i] - mean; s2 += d * d; }
  const float rstd = rsqrtf(wave_sum(s2) * (1.0f / D_DIM) + 1e-5f);
  bf16* q = dst + row * D_DIM;
#pragma unroll
  for (int i = 0; i < 8; ++i) q[i * 32 + lane] = (bf16)((v[i] - mean) * rstd);
}

// v += LN(dv); refresh bf16 copy and bf16-transposed copy
__global__ __launch_bounds__(128) void k_res_ln(const float* __restrict__ dv,
                                                float* __restrict__ vf,
                                                bf16* __restrict__ vb,
                                                bf16* __restrict__ vbT) {
  const int lane = threadIdx.x & 31;
  const long row = (long)blockIdx.x * 4 + (threadIdx.x >> 5);
  const float* p = dv + row * D_DIM;
  float v[8]; float s = 0.f;
#pragma unroll
  for (int i = 0; i < 8; ++i) { v[i] = p[i * 32 + lane]; s += v[i]; }
  const float mean = wave_sum(s) * (1.0f / D_DIM);
  float s2 = 0.f;
#pragma unroll
  for (int i = 0; i < 8; ++i) { float d = v[i] - mean; s2 += d * d; }
  const float rstd = rsqrtf(wave_sum(s2) * (1.0f / D_DIM) + 1e-5f);
#pragma unroll
  for (int i = 0; i < 8; ++i) {
    const int dcol = i * 32 + lane;
    float nv = vf[row * D_DIM + dcol] + (v[i] - mean) * rstd;
    vf[row * D_DIM + dcol] = nv;
    vb[row * D_DIM + dcol] = (bf16)nv;
    vbT[(long)dcol * T_SEQ + row] = (bf16)nv;
  }
}

// ---------------- fused epilogue helpers ----------------
__device__ __forceinline__ void relu_rope_store(v8f acc, int m0, int hi, int n,
                                                bf16* __restrict__ xo,
                                                bf16* __restrict__ qo) {
  const float freq = exp2f(-(float)(n & ~1) * (1.0f / 512.0f)) * INV_2PI;
#pragma unroll
  for (int j = 0; j < 8; ++j) {
    const int t = m0 + j + 8 * hi;
    float xv = fmaxf(acc[j], 0.f);                 // relu before rope
    float xp = __shfl_xor(xv, 1, 32);              // pair partner column n^1
    float rot = (n & 1) ? xp : -xp;                // (-x[n+1], x[n]) pattern
    float ph = (float)t * freq;
    ph = (ph - floorf(ph)) * TWO_PI;
    float sn, cs;
    __sincosf(ph, &sn, &cs);
    const long o = (long)t * NH + n;
    xo[o] = (bf16)xv;
    qo[o] = (bf16)(xv * cs + rot * sn);
  }
}

__device__ __forceinline__ void gate_store(v8f acc, int m0, int hi, int n,
                                           bf16* __restrict__ xy) {
#pragma unroll
  for (int j = 0; j < 8; ++j) {
    const int t = m0 + j + 8 * hi;
    const long o = (long)t * NH + n;
    float g = fmaxf(acc[j], 0.f);
    xy[o] = (bf16)(g * (float)xy[o]);              // y = relu(.) * x, in place
  }
}

// ---------------- GEMM 1: x = relu(v . Wx[h]);  qr = rope(x) ----------------
// M=T, K=D, N=NH per head; 2x2 tiles (32x32) per wave.
__global__ __launch_bounds__(128) void k_gemm_x_rope(const bf16* __restrict__ vb,   // (T,D)
                                                     const bf16* __restrict__ WxT,  // (H,NH,D)
                                                     bf16* __restrict__ x_out,      // (H,T,NH)
                                                     bf16* __restrict__ qr_out) {   // (H,T,NH)
  const int lane = threadIdx.x & 31;
  const int wv   = threadIdx.x >> 5;
  const int m0 = blockIdx.x * 32;
  const int n0 = (blockIdx.y * 4 + wv) * 32;
  const int h  = blockIdx.z;
  const bf16* W = WxT + (long)h * NH * D_DIM;
  v8f a00 = {}, a01 = {}, a10 = {}, a11 = {};
  for (int k0 = 0; k0 < D_DIM; k0 += 32) {
    v16bf A0 = load_frag(vb, D_DIM, m0, k0, lane);
    v16bf A1 = load_frag(vb, D_DIM, m0 + 16, k0, lane);
    v16bf B0 = load_frag(W, D_DIM, n0, k0, lane);
    v16bf B1 = load_frag(W, D_DIM, n0 + 16, k0, lane);
    a00 = wmma_bf16(A0, B0, a00);
    a01 = wmma_bf16(A0, B1, a01);
    a10 = wmma_bf16(A1, B0, a10);
    a11 = wmma_bf16(A1, B1, a11);
  }
  const int c  = lane & 15;
  const int hi = lane >> 4;
  bf16* xo = x_out + (long)h * T_SEQ * NH;
  bf16* qo = qr_out + (long)h * T_SEQ * NH;
  relu_rope_store(a00, m0,      hi, n0 + c,      xo, qo);
  relu_rope_store(a01, m0,      hi, n0 + 16 + c, xo, qo);
  relu_rope_store(a10, m0 + 16, hi, n0 + c,      xo, qo);
  relu_rope_store(a11, m0 + 16, hi, n0 + 16 + c, xo, qo);
}

// ---------------- GEMM 2: scores = qr . qr^T, strictly causal ----------------
// 64x64 output block per workgroup; qr panels staged in double-buffered LDS;
// each of 4 waves computes a 32x32 register tile (2x2 WMMA accumulators).
__global__ __launch_bounds__(128) void k_scores(const bf16* __restrict__ qr,  // (H,T,NH)
                                                bf16* __restrict__ sc) {      // (H,T,T)
  __shared__ __align__(16) bf16 Apan[2][64 * LDS_STRIDE];
  __shared__ __align__(16) bf16 Bpan[2][64 * LDS_STRIDE];
  const int tid  = threadIdx.x;
  const int lane = tid & 31;
  const int wv   = tid >> 5;
  const int tBlk = blockIdx.x, sBlk = blockIdx.y, h = blockIdx.z;
  const int tBase = tBlk * 64, sBase = sBlk * 64;
  bf16* out = sc + (long)h * T_SEQ * T_SEQ;

  if (sBlk > tBlk) {                       // fully above diagonal: vector zero-fill
    v8bf z = {};
#pragma unroll
    for (int i = 0; i < 4; ++i) {
      const int idx = tid + i * 128;       // 512 chunks of 8 elems = 64x64
      const int row = idx >> 3, q = idx & 7;
      *(v8bf*)(out + (long)(tBase + row) * T_SEQ + sBase + q * 8) = z;
    }
    return;
  }

  const bf16* Q = qr + (long)h * T_SEQ * NH;
  const int frow = tid >> 2, fq = tid & 3;           // each thread: 2 rows per panel

  // fill one K-chunk (64 rows x 32 K) of both panels; coalesced b128 loads
  auto fill = [&](int buf, int k0) {
#pragma unroll
    for (int i = 0; i < 2; ++i) {
      const int row = frow + i * 32;
      const bf16* ga = Q + (long)(tBase + row) * NH + k0 + fq * 8;
      const bf16* gb = Q + (long)(sBase + row) * NH + k0 + fq * 8;
      if (k0 + 32 < NH) {
        __builtin_prefetch((const void*)(ga + 32), 0, 0);
        __builtin_prefetch((const void*)(gb + 32), 0, 0);
      }
      *(v8bf*)&Apan[buf][row * LDS_STRIDE + fq * 8] = *(const v8bf*)ga;
      *(v8bf*)&Bpan[buf][row * LDS_STRIDE + fq * 8] = *(const v8bf*)gb;
    }
  };

  const int wt = (wv >> 1) * 32;                     // wave's 32-row strip in block
  const int ws = (wv & 1) * 32;                      // wave's 32-col strip in block
  v8f a00 = {}, a01 = {}, a10 = {}, a11 = {};

  fill(0, 0);
  __syncthreads();
  for (int k0 = 0; k0 < NH; k0 += 32) {
    const int buf = (k0 >> 5) & 1;
    if (k0 + 32 < NH) fill(buf ^ 1, k0 + 32);
    v16bf A0 = lds_frag(Apan[buf], wt, lane);
    v16bf A1 = lds_frag(Apan[buf], wt + 16, lane);
    v16bf B0 = lds_frag(Bpan[buf], ws, lane);
    v16bf B1 = lds_frag(Bpan[buf], ws + 16, lane);
    a00 = wmma_bf16(A0, B0, a00);
    a01 = wmma_bf16(A0, B1, a01);
    a10 = wmma_bf16(A1, B0, a10);
    a11 = wmma_bf16(A1, B1, a11);
    __syncthreads();
  }

  const int c  = lane & 15;
  const int hi = lane >> 4;
  auto store_tile = [&](v8f acc, int tOff, int sOff) {
    const int s = sBase + sOff + c;
#pragma unroll
    for (int j = 0; j < 8; ++j) {
      const int t = tBase + tOff + j + 8 * hi;
      float v = (s < t) ? acc[j] : 0.f;              // strict causal mask
      out[(long)t * T_SEQ + s] = (bf16)v;
    }
  };
  store_tile(a00, wt,      ws);
  store_tile(a01, wt,      ws + 16);
  store_tile(a10, wt + 16, ws);
  store_tile(a11, wt + 16, ws + 16);
}

// ---------------- GEMM 3: a = scores . v   (K truncated at diagonal) ----------------
__global__ __launch_bounds__(128) void k_attnV(const bf16* __restrict__ sc,   // (H,T,T)
                                               const bf16* __restrict__ vbT,  // (D,T)
                                               float* __restrict__ a_out) {   // (H,T,D)
  const int lane = threadIdx.x & 31;
  const int wv   = threadIdx.x >> 5;
  const int t0 = blockIdx.x * 16;
  const int n0 = (blockIdx.y * 4 + wv) * 16;
  const int h  = blockIdx.z;
  const bf16* S = sc + (long)h * T_SEQ * T_SEQ;
  v8f acc = {};
  const int kend = t0 + 16;                          // scores zero for s >= t
  for (int k0 = 0; k0 < kend; k0 += 32) {
    v16bf a = load_frag(S, T_SEQ, t0, k0, lane);
    v16bf b = load_frag(vbT, T_SEQ, n0, k0, lane);   // v^T: contiguous B frag
    acc = wmma_bf16(a, b, acc);
  }
  const int c  = lane & 15;
  const int hi = lane >> 4;
#pragma unroll
  for (int j = 0; j < 8; ++j) {
    const int t = t0 + j + 8 * hi;
    a_out[((long)h * T_SEQ + t) * D_DIM + (n0 + c)] = acc[j];
  }
}

// ---------------- GEMM 4: y = relu(LN(a) . Wy[h]) * x  (overwrites x in place) -------
__global__ __launch_bounds__(128) void k_gemm_y(const bf16* __restrict__ aln,  // (H,T,D)
                                                const bf16* __restrict__ WyT,  // (H,NH,D)
                                                bf16* __restrict__ xy) {       // (H,T,NH)
  const int lane = threadIdx.x & 31;
  const int wv   = threadIdx.x >> 5;
  const int m0 = blockIdx.x * 32;
  const int n0 = (blockIdx.y * 4 + wv) * 32;
  const int h  = blockIdx.z;
  const bf16* A = aln + (long)h * T_SEQ * D_DIM;
  const bf16* W = WyT + (long)h * NH * D_DIM;
  v8f a00 = {}, a01 = {}, a10 = {}, a11 = {};
  for (int k0 = 0; k0 < D_DIM; k0 += 32) {
    v16bf A0 = load_frag(A, D_DIM, m0, k0, lane);
    v16bf A1 = load_frag(A, D_DIM, m0 + 16, k0, lane);
    v16bf B0 = load_frag(W, D_DIM, n0, k0, lane);
    v16bf B1 = load_frag(W, D_DIM, n0 + 16, k0, lane);
    a00 = wmma_bf16(A0, B0, a00);
    a01 = wmma_bf16(A0, B1, a01);
    a10 = wmma_bf16(A1, B0, a10);
    a11 = wmma_bf16(A1, B1, a11);
  }
  const int c  = lane & 15;
  const int hi = lane >> 4;
  bf16* xh = xy + (long)h * T_SEQ * NH;
  gate_store(a00, m0,      hi, n0 + c,      xh);
  gate_store(a01, m0,      hi, n0 + 16 + c, xh);
  gate_store(a10, m0 + 16, hi, n0 + c,      xh);
  gate_store(a11, m0 + 16, hi, n0 + 16 + c, xh);
}

// ---------------- GEMM 5: dv = y . encoder, split-K with f32 atomics ----------------
__global__ __launch_bounds__(128) void k_gemm_enc(const bf16* __restrict__ y,     // (H,T,NH)
                                                  const bf16* __restrict__ encT,  // (D,N)
                                                  float* __restrict__ dv) {       // (T,D)
  const int lane = threadIdx.x & 31;
  const int wv   = threadIdx.x >> 5;
  const int m0 = blockIdx.x * 32;
  const int n0 = (blockIdx.y * 4 + wv) * 32;
  const int kbeg = blockIdx.z * (N_DIM / 8);
  const int kend = kbeg + (N_DIM / 8);
  v8f a00 = {}, a01 = {}, a10 = {}, a11 = {};
  for (int k0 = kbeg; k0 < kend; k0 += 32) {
    const int h  = k0 >> 13;                         // k0 / NH
    const int kk = k0 & (NH - 1);
    const bf16* Y = y + (long)h * T_SEQ * NH;
    v16bf A0 = load_frag(Y, NH, m0, kk, lane);
    v16bf A1 = load_frag(Y, NH, m0 + 16, kk, lane);
    v16bf B0 = load_frag(encT, N_DIM, n0, k0, lane);
    v16bf B1 = load_frag(encT, N_DIM, n0 + 16, k0, lane);
    a00 = wmma_bf16(A0, B0, a00);
    a01 = wmma_bf16(A0, B1, a01);
    a10 = wmma_bf16(A1, B0, a10);
    a11 = wmma_bf16(A1, B1, a11);
  }
  const int c  = lane & 15;
  const int hi = lane >> 4;
#pragma unroll
  for (int j = 0; j < 8; ++j) {
    const int t0 = m0 + j + 8 * hi;
    atomicAdd(&dv[(long)t0 * D_DIM + (n0 + c)],            a00[j]);
    atomicAdd(&dv[(long)t0 * D_DIM + (n0 + 16 + c)],       a01[j]);
    atomicAdd(&dv[(long)(t0 + 16) * D_DIM + (n0 + c)],     a10[j]);
    atomicAdd(&dv[(long)(t0 + 16) * D_DIM + (n0 + 16 + c)], a11[j]);
  }
}

// ---------------- GEMM 6: logits = LN(v) . readout (f32 out) ----------------
__global__ __launch_bounds__(128) void k_logits(const bf16* __restrict__ lnv,  // (T,D)
                                                const bf16* __restrict__ roT,  // (V,D)
                                                float* __restrict__ out) {     // (T,V)
  const int lane = threadIdx.x & 31;
  const int wv   = threadIdx.x >> 5;
  const int m0 = blockIdx.x * 16;
  const int n0 = (blockIdx.y * 4 + wv) * 16;
  v8f acc = {};
  for (int k0 = 0; k0 < D_DIM; k0 += 32) {
    v16bf a = load_frag(lnv, D_DIM, m0, k0, lane);
    v16bf b = load_frag(roT, D_DIM, n0, k0, lane);
    acc = wmma_bf16(a, b, acc);
  }
  const int c  = lane & 15;
  const int hi = lane >> 4;
#pragma unroll
  for (int j = 0; j < 8; ++j) {
    const int t = m0 + j + 8 * hi;
    out[(long)t * V_DIM + (n0 + c)] = acc[j];
  }
}

// ---------------- host orchestration ----------------
extern "C" void kernel_launch(void* const* d_in, const int* in_sizes, int n_in,
                              void* d_out, int out_size, void* d_ws, size_t ws_size,
                              hipStream_t stream) {
  (void)in_sizes; (void)n_in; (void)out_size; (void)ws_size;
  const int*   idx = (const int*)d_in[0];
  const float* wte = (const float*)d_in[1];
  const float* dx  = (const float*)d_in[2];
  const float* dy  = (const float*)d_in[3];
  const float* enc = (const float*)d_in[4];
  const float* ro  = (const float*)d_in[5];
  float* out = (float*)d_out;

  // workspace carving (256B aligned); total ~204 MB
  size_t off = 0;
  auto take = [&](size_t bytes) -> char* {
    char* p = (char*)d_ws + off;
    off += (bytes + 255) & ~(size_t)255;
    return p;
  };
  bf16*  WxT  = (bf16*)take(sizeof(bf16) * (size_t)H_HEADS * NH * D_DIM);
  bf16*  WyT  = (bf16*)take(sizeof(bf16) * (size_t)H_HEADS * NH * D_DIM);
  bf16*  EncT = (bf16*)take(sizeof(bf16) * (size_t)D_DIM * N_DIM);
  bf16*  RoT  = (bf16*)take(sizeof(bf16) * (size_t)V_DIM * D_DIM);
  float* vf   = (float*)take(sizeof(float) * (size_t)T_SEQ * D_DIM);
  bf16*  vb   = (bf16*)take(sizeof(bf16) * (size_t)T_SEQ * D_DIM);
  bf16*  vbT  = (bf16*)take(sizeof(bf16) * (size_t)D_DIM * T_SEQ);
  bf16*  xb   = (bf16*)take(sizeof(bf16) * (size_t)H_HEADS * T_SEQ * NH);  // x, then y
  bf16*  qrb  = (bf16*)take(sizeof(bf16) * (size_t)H_HEADS * T_SEQ * NH);
  bf16*  scb  = (bf16*)take(sizeof(bf16) * (size_t)H_HEADS * T_SEQ * T_SEQ);
  float* af   = (float*)take(sizeof(float) * (size_t)H_HEADS * T_SEQ * D_DIM);
  bf16*  alnb = (bf16*)take(sizeof(bf16) * (size_t)H_HEADS * T_SEQ * D_DIM);
  float* dvf  = (float*)take(sizeof(float) * (size_t)T_SEQ * D_DIM);
  bf16*  lnvb = (bf16*)take(sizeof(bf16) * (size_t)T_SEQ * D_DIM);

  // weight convert + transpose (one-time per launch)
  {
    long pb = (long)D_DIM * NH;                       // per-head
    k_cvt_T<<<dim3((unsigned)((pb + 255) / 256), H_HEADS), dim3(256), 0, stream>>>(
        dx, WxT, D_DIM, NH, pb);
    k_cvt_T<<<dim3((unsigned)((pb + 255) / 256), H_HEADS), dim3(256), 0, stream>>>(
        dy, WyT, D_DIM, NH, pb);
    long pe = (long)N_DIM * D_DIM;
    k_cvt_T<<<dim3((unsigned)((pe + 255) / 256), 1), dim3(256), 0, stream>>>(
        enc, EncT, N_DIM, D_DIM, pe);
    long pr = (long)D_DIM * V_DIM;
    k_cvt_T<<<dim3((unsigned)((pr + 255) / 256), 1), dim3(256), 0, stream>>>(
        ro, RoT, D_DIM, V_DIM, pr);
  }

  k_embed<<<dim3(T_SEQ / 4), dim3(128), 0, stream>>>(idx, wte, vf, vb, vbT);

  for (int l = 0; l < L_LAYERS; ++l) {
    k_gemm_x_rope<<<dim3(32, 64, 4), dim3(128), 0, stream>>>(vb, WxT, xb, qrb);
    k_scores<<<dim3(16, 16, 4), dim3(128), 0, stream>>>(qrb, scb);
    k_attnV<<<dim3(64, 4, 4), dim3(128), 0, stream>>>(scb, vbT, af);
    k_ln_rows<<<dim3((H_HEADS * T_SEQ) / 4), dim3(128), 0, stream>>>(af, alnb);
    k_gemm_y<<<dim3(32, 64, 4), dim3(128), 0, stream>>>(alnb, WyT, xb);
    hipMemsetAsync(dvf, 0, sizeof(float) * (size_t)T_SEQ * D_DIM, stream);
    k_gemm_enc<<<dim3(32, 2, 8), dim3(128), 0, stream>>>(xb, EncT, dvf);
    k_res_ln<<<dim3(T_SEQ / 4), dim3(128), 0, stream>>>(dvf, vf, vb, vbT);
  }

  k_ln_rows<<<dim3(T_SEQ / 4), dim3(128), 0, stream>>>(vf, lnvb);
  k_logits<<<dim3(64, 4), dim3(128), 0, stream>>>(lnvb, RoT, out);
}
